// AffineCouplingLayer_77953656423056
// MI455X (gfx1250) — matrix operand order repaired and checked by reading, not compiled, not run on hardware
//
#include <hip/hip_runtime.h>

#define B_TOTAL 65536
#define DIM     128
#define D_IN    64
#define HID     512
#define LOG_CLAMP 2.0f

typedef __attribute__((ext_vector_type(16))) _Float16 v16h;
typedef __attribute__((ext_vector_type(8)))  float    v8f;

union F16Frag {
  v16h     v;
  uint4    q[2];
  _Float16 h[16];
};

static __device__ __forceinline__ v8f wmma_f16(v16h a, v16h b, v8f c) {
  // D = A(16x32 f16) x B(32x16 f16) + C(16x16 f32)
  return __builtin_amdgcn_wmma_f32_16x16x32_f16(false, a, false, b, (short)0, c,
                                                false, false);
}

static __device__ __forceinline__ float relu(float v) { return v > 0.f ? v : 0.f; }

// ---------------------------------------------------------------------------
// B fragment (32x16 KxN tile of W^T), ISA 7.12.4 layout:
// lane L -> n = nt*16 + L%16 ; halfs e -> k = ks*32 + (L>=16?8:0) + (e&7) + (e>>3)*16.
// PACKED: fragments pre-converted to f16 and laid out contiguously so each
// lane does two aligned b128 loads (frag = 1KB block, lane L owns 32B).
// Fallback: two runs of 8 consecutive fp32 of W[n][*] (k0.., k0+16..) -> 4x float4 + cvt.
// ---------------------------------------------------------------------------
template<bool PACKED, int K>
static __device__ __forceinline__ v16h load_b_frag(const void* __restrict__ Ws,
                                                   int nt, int ks,
                                                   int n16, int hi, int lane) {
  F16Frag r;
  if constexpr (PACKED) {
    const _Float16* P = (const _Float16*)Ws;
    const uint4* p =
        (const uint4*)(P + (((size_t)(nt * (K / 32) + ks)) << 9) + lane * 16);
    r.q[0] = p[0];
    r.q[1] = p[1];
  } else {
    const float* W = (const float*)Ws;
    int n = nt * 16 + n16;
    const float4* p = (const float4*)(W + (size_t)n * K + ks * 32 + hi * 8);
    float4 f0 = p[0], f1 = p[1], f2 = p[4], f3 = p[5];
    r.h[0]  = (_Float16)f0.x; r.h[1]  = (_Float16)f0.y;
    r.h[2]  = (_Float16)f0.z; r.h[3]  = (_Float16)f0.w;
    r.h[4]  = (_Float16)f1.x; r.h[5]  = (_Float16)f1.y;
    r.h[6]  = (_Float16)f1.z; r.h[7]  = (_Float16)f1.w;
    r.h[8]  = (_Float16)f2.x; r.h[9]  = (_Float16)f2.y;
    r.h[10] = (_Float16)f2.z; r.h[11] = (_Float16)f2.w;
    r.h[12] = (_Float16)f3.x; r.h[13] = (_Float16)f3.y;
    r.h[14] = (_Float16)f3.z; r.h[15] = (_Float16)f3.w;
  }
  return r.v;
}

// ---------------------------------------------------------------------------
// Activations in LDS: h[m][k] f16, XOR-swizzled on 8-half chunks:
// (m,k) stored at m*512 + ((k/8)^(m&15))*8 + k%8  -> A-frag b128 loads hit
// 64 distinct banks across the wave.
// A fragment (16x32, ISA 7.12.2): lane L row m=t*16+L%16,
// halfs [k0..k0+7, k0+16..k0+23], k0 = ks*32 + (L>=16)*8.
// ---------------------------------------------------------------------------
static __device__ __forceinline__ v16h load_a_frag(const _Float16* __restrict__ buf,
                                                   int t, int ks, int m16, int hi) {
  int m  = t * 16 + m16;
  int c0 = ((ks * 4 + hi)     ^ m16) << 3;
  int c1 = ((ks * 4 + 2 + hi) ^ m16) << 3;
  F16Frag r;
  r.q[0] = *(const uint4*)(buf + m * HID + c0);
  r.q[1] = *(const uint4*)(buf + m * HID + c1);
  return r.v;
}

static __device__ __forceinline__ void store_h(_Float16* __restrict__ buf,
                                               int m, int n, float v) {
  int chunk = (n >> 3) ^ (m & 15);
  buf[m * HID + (chunk << 3) + (n & 7)] = (_Float16)v;
}

// ---------------------------------------------------------------------------
// Layers: 32 rows/wave as two 16-row M-tiles; two N-tiles per iteration ->
// four independent WMMA accumulation chains for XDL pipelining.
// ---------------------------------------------------------------------------
template<bool PACKED>
static __device__ __forceinline__ void layer_in(const void* __restrict__ Ws,
                                                const float* __restrict__ b,
                                                const v16h areg[2][2],
                                                _Float16* __restrict__ dst,
                                                int n16, int hi, int lane) {
  for (int nt = 0; nt < HID / 16; nt += 2) {
    float bv0 = b[nt * 16 + n16];
    float bv1 = b[nt * 16 + 16 + n16];
    v8f a00, a10, a01, a11;
#pragma unroll
    for (int i = 0; i < 8; ++i) { a00[i] = bv0; a10[i] = bv0; a01[i] = bv1; a11[i] = bv1; }
#pragma unroll
    for (int ks = 0; ks < D_IN / 32; ++ks) {
      v16h bf0 = load_b_frag<PACKED, D_IN>(Ws, nt,     ks, n16, hi, lane);
      v16h bf1 = load_b_frag<PACKED, D_IN>(Ws, nt + 1, ks, n16, hi, lane);
      a00 = wmma_f16(areg[0][ks], bf0, a00);
      a10 = wmma_f16(areg[1][ks], bf0, a10);
      a01 = wmma_f16(areg[0][ks], bf1, a01);
      a11 = wmma_f16(areg[1][ks], bf1, a11);
    }
#pragma unroll
    for (int r = 0; r < 8; ++r) {
      store_h(dst, r + hi * 8,      nt * 16 + n16,      relu(a00[r]));
      store_h(dst, 16 + r + hi * 8, nt * 16 + n16,      relu(a10[r]));
      store_h(dst, r + hi * 8,      nt * 16 + 16 + n16, relu(a01[r]));
      store_h(dst, 16 + r + hi * 8, nt * 16 + 16 + n16, relu(a11[r]));
    }
  }
}

template<bool PACKED>
static __device__ __forceinline__ void layer_hid(const void* __restrict__ Ws,
                                                 const float* __restrict__ b,
                                                 const _Float16* __restrict__ src,
                                                 _Float16* __restrict__ dst,
                                                 int m16, int n16, int hi, int lane) {
  for (int nt = 0; nt < HID / 16; nt += 2) {
    float bv0 = b[nt * 16 + n16];
    float bv1 = b[nt * 16 + 16 + n16];
    v8f a00, a10, a01, a11;
#pragma unroll
    for (int i = 0; i < 8; ++i) { a00[i] = bv0; a10[i] = bv0; a01[i] = bv1; a11[i] = bv1; }
    for (int ks = 0; ks < HID / 32; ++ks) {
      v16h bf0 = load_b_frag<PACKED, HID>(Ws, nt,     ks, n16, hi, lane);
      v16h bf1 = load_b_frag<PACKED, HID>(Ws, nt + 1, ks, n16, hi, lane);
      v16h af0 = load_a_frag(src, 0, ks, m16, hi);
      v16h af1 = load_a_frag(src, 1, ks, m16, hi);
      a00 = wmma_f16(af0, bf0, a00);
      a10 = wmma_f16(af1, bf0, a10);
      a01 = wmma_f16(af0, bf1, a01);
      a11 = wmma_f16(af1, bf1, a11);
    }
#pragma unroll
    for (int r = 0; r < 8; ++r) {
      store_h(dst, r + hi * 8,      nt * 16 + n16,      relu(a00[r]));
      store_h(dst, 16 + r + hi * 8, nt * 16 + n16,      relu(a10[r]));
      store_h(dst, r + hi * 8,      nt * 16 + 16 + n16, relu(a01[r]));
      store_h(dst, 16 + r + hi * 8, nt * 16 + 16 + n16, relu(a11[r]));
    }
  }
}

template<bool PACKED>
static __device__ __forceinline__ void layer_out(const void* __restrict__ Ws,
                                                 const float* __restrict__ b,
                                                 const _Float16* __restrict__ src,
                                                 v8f out[4][2],
                                                 int m16, int n16, int hi, int lane) {
#pragma unroll
  for (int nt = 0; nt < 4; nt += 2) {
    float bv0 = b[nt * 16 + n16];
    float bv1 = b[nt * 16 + 16 + n16];
    v8f a00, a10, a01, a11;
#pragma unroll
    for (int i = 0; i < 8; ++i) { a00[i] = bv0; a10[i] = bv0; a01[i] = bv1; a11[i] = bv1; }
    for (int ks = 0; ks < HID / 32; ++ks) {
      v16h bf0 = load_b_frag<PACKED, HID>(Ws, nt,     ks, n16, hi, lane);
      v16h bf1 = load_b_frag<PACKED, HID>(Ws, nt + 1, ks, n16, hi, lane);
      v16h af0 = load_a_frag(src, 0, ks, m16, hi);
      v16h af1 = load_a_frag(src, 1, ks, m16, hi);
      a00 = wmma_f16(af0, bf0, a00);
      a10 = wmma_f16(af1, bf0, a10);
      a01 = wmma_f16(af0, bf1, a01);
      a11 = wmma_f16(af1, bf1, a11);
    }
    out[nt][0] = a00; out[nt][1] = a10;
    out[nt + 1][0] = a01; out[nt + 1][1] = a11;
  }
}

template<bool PACKED>
static __device__ __forceinline__ void run_mlp(
    const void* Win, const float* bin,
    const void* Wh0, const void* Wh1, const float* bhid,
    const void* Wout, const float* bout,
    const v16h areg[2][2], _Float16* buf0, _Float16* buf1,
    v8f out[4][2], int m16, int n16, int hi, int lane) {
  layer_in <PACKED>(Win, bin,        areg, buf0, n16, hi, lane);
  layer_hid<PACKED>(Wh0, bhid,       buf0, buf1, m16, n16, hi, lane);
  layer_hid<PACKED>(Wh1, bhid + HID, buf1, buf0, m16, n16, hi, lane);
  layer_out<PACKED>(Wout, bout,      buf0, out,  m16, n16, hi, lane);
}

// ---------------------------------------------------------------------------
// Weight prepack: fp32 [N][K] -> f16 B-fragments, contiguous frag-major.
// dst[tid]: tid = (nt*(K/32)+ks)*512 + lane*16 + e
//           value = W[(nt*16 + lane%16) * K + ks*32 + (lane/16)*8 + (e&7) + (e>>3)*16]
// ---------------------------------------------------------------------------
__global__ __launch_bounds__(256)
void prepack_kernel(const float* __restrict__ W, _Float16* __restrict__ dst,
                    int K, int lgKS, int total) {
  int tid = blockIdx.x * 256 + threadIdx.x;
  if (tid >= total) return;
  int f    = tid >> 9;
  int r    = tid & 511;
  int lane = r >> 4;
  int e    = r & 15;
  int nt   = f >> lgKS;
  int ks   = f & ((1 << lgKS) - 1);
  int n    = nt * 16 + (lane & 15);
  int k    = ks * 32 + (lane >> 4) * 8 + (e & 7) + (e >> 3) * 16;
  dst[tid] = (_Float16)W[(size_t)n * K + k];
}

// ---------------------------------------------------------------------------
// Main kernel: one wave (32 threads) per workgroup, 32 batch rows per wave.
// ---------------------------------------------------------------------------
template<bool PACKED>
__global__ __launch_bounds__(32)
void affine_coupling_kernel(
    const float* __restrict__ z,
    const void* sWin, const float* sbin, const void* sWh0, const void* sWh1,
    const float* sbhid, const void* sWout, const float* sbout,
    const void* tWin, const float* tbin, const void* tWh0, const void* tWh1,
    const float* tbhid, const void* tWout, const float* tbout,
    float* __restrict__ out, float* __restrict__ logdet) {
  __shared__ _Float16 buf0[32 * HID];
  __shared__ _Float16 buf1[32 * HID];

  const int lane = threadIdx.x;
  const int m16  = lane & 15;
  const int hi   = lane >> 4;
  const int n16  = m16;
  const size_t base = (size_t)blockIdx.x * 32;

  // z1 passthrough: out[:, :64] = z[:, :64]
  for (int i = lane; i < 32 * 16; i += 32) {
    int r = i >> 4, c = i & 15;
    ((float4*)(out + (base + r) * DIM))[c] =
        ((const float4*)(z + (base + r) * DIM))[c];
  }

  // Layer-0 A fragments from global z (K = 64 -> 2 k-steps, 2 row tiles).
  v16h areg[2][2];
#pragma unroll
  for (int t = 0; t < 2; ++t) {
#pragma unroll
    for (int ks = 0; ks < 2; ++ks) {
      const float* zr = z + (base + t * 16 + m16) * DIM;
      int k0 = ks * 32 + hi * 8;
      float4 f0 = *(const float4*)(zr + k0);
      float4 f1 = *(const float4*)(zr + k0 + 4);
      float4 f2 = *(const float4*)(zr + k0 + 16);
      float4 f3 = *(const float4*)(zr + k0 + 20);
      F16Frag fr;
      fr.h[0]  = (_Float16)f0.x; fr.h[1]  = (_Float16)f0.y;
      fr.h[2]  = (_Float16)f0.z; fr.h[3]  = (_Float16)f0.w;
      fr.h[4]  = (_Float16)f1.x; fr.h[5]  = (_Float16)f1.y;
      fr.h[6]  = (_Float16)f1.z; fr.h[7]  = (_Float16)f1.w;
      fr.h[8]  = (_Float16)f2.x; fr.h[9]  = (_Float16)f2.y;
      fr.h[10] = (_Float16)f2.z; fr.h[11] = (_Float16)f2.w;
      fr.h[12] = (_Float16)f3.x; fr.h[13] = (_Float16)f3.y;
      fr.h[14] = (_Float16)f3.z; fr.h[15] = (_Float16)f3.w;
      areg[t][ks] = fr.v;
    }
  }

  // Scale MLP -> clamped log_s held in registers across the t-MLP.
  v8f slog[4][2];
  run_mlp<PACKED>(sWin, sbin, sWh0, sWh1, sbhid, sWout, sbout,
                  areg, buf0, buf1, slog, m16, n16, hi, lane);
#pragma unroll
  for (int nt = 0; nt < 4; ++nt)
#pragma unroll
    for (int t = 0; t < 2; ++t)
#pragma unroll
      for (int r = 0; r < 8; ++r) {
        float c = slog[nt][t][r];
        c = c > LOG_CLAMP ? LOG_CLAMP : c;
        c = c < -LOG_CLAMP ? -LOG_CLAMP : c;
        slog[nt][t][r] = c;
      }

  // Translate MLP.
  v8f tacc[4][2];
  run_mlp<PACKED>(tWin, tbin, tWh0, tWh1, tbhid, tWout, tbout,
                  areg, buf0, buf1, tacc, m16, n16, hi, lane);

  // Epilogue: z2_out = z2*exp(log_s) + t ; log_det = sum_n log_s.
#pragma unroll
  for (int t = 0; t < 2; ++t) {
#pragma unroll
    for (int r = 0; r < 8; ++r) {
      size_t row = base + t * 16 + r + hi * 8;
      float ldsum = 0.f;
#pragma unroll
      for (int nt = 0; nt < 4; ++nt) {
        int n = nt * 16 + n16;
        float ls = slog[nt][t][r];
        float z2 = z[row * DIM + D_IN + n];
        out[row * DIM + D_IN + n] = z2 * __expf(ls) + tacc[nt][t][r];
        ldsum += ls;
      }
      ldsum += __shfl_xor(ldsum, 1, 32);
      ldsum += __shfl_xor(ldsum, 2, 32);
      ldsum += __shfl_xor(ldsum, 4, 32);
      ldsum += __shfl_xor(ldsum, 8, 32);
      if (n16 == 0) logdet[row] = ldsum;
    }
  }
}

extern "C" void kernel_launch(void* const* d_in, const int* in_sizes, int n_in,
                              void* d_out, int out_size, void* d_ws, size_t ws_size,
                              hipStream_t stream) {
  (void)in_sizes; (void)n_in; (void)out_size;
  const float* z     = (const float*)d_in[0];
  const float* sWin  = (const float*)d_in[1];
  const float* sbin  = (const float*)d_in[2];
  const float* sWhid = (const float*)d_in[3];
  const float* sbhid = (const float*)d_in[4];
  const float* sWout = (const float*)d_in[5];
  const float* sbout = (const float*)d_in[6];
  const float* tWin  = (const float*)d_in[7];
  const float* tbin  = (const float*)d_in[8];
  const float* tWhid = (const float*)d_in[9];
  const float* tbhid = (const float*)d_in[10];
  const float* tWout = (const float*)d_in[11];
  const float* tbout = (const float*)d_in[12];

  float* out    = (float*)d_out;
  float* logdet = out + (size_t)B_TOTAL * DIM;

  // Packed-weight sizes (halfs)
  const size_t SZ_IN  = (size_t)HID * D_IN;   // 32768
  const size_t SZ_HID = (size_t)HID * HID;    // 262144
  const size_t SZ_OUT = (size_t)D_IN * HID;   // 32768
  const size_t TOTAL_HALFS = 2 * (SZ_IN + 2 * SZ_HID + SZ_OUT);  // 1,179,648

  dim3 grid(B_TOTAL / 32);

  if (ws_size >= TOTAL_HALFS * sizeof(_Float16)) {
    _Float16* w = (_Float16*)d_ws;
    _Float16* p_sWin  = w;
    _Float16* p_sWh0  = p_sWin + SZ_IN;
    _Float16* p_sWh1  = p_sWh0 + SZ_HID;
    _Float16* p_sWout = p_sWh1 + SZ_HID;
    _Float16* p_tWin  = p_sWout + SZ_OUT;
    _Float16* p_tWh0  = p_tWin + SZ_IN;
    _Float16* p_tWh1  = p_tWh0 + SZ_HID;
    _Float16* p_tWout = p_tWh1 + SZ_HID;

    // Prepack all 8 weight matrices (stream-ordered before the main kernel).
    prepack_kernel<<<(int)(SZ_IN  / 256), 256, 0, stream>>>(sWin,            p_sWin,  D_IN, 1, (int)SZ_IN);
    prepack_kernel<<<(int)(SZ_HID / 256), 256, 0, stream>>>(sWhid,           p_sWh0,  HID,  4, (int)SZ_HID);
    prepack_kernel<<<(int)(SZ_HID / 256), 256, 0, stream>>>(sWhid + SZ_HID,  p_sWh1,  HID,  4, (int)SZ_HID);
    prepack_kernel<<<(int)(SZ_OUT / 256), 256, 0, stream>>>(sWout,           p_sWout, HID,  4, (int)SZ_OUT);
    prepack_kernel<<<(int)(SZ_IN  / 256), 256, 0, stream>>>(tWin,            p_tWin,  D_IN, 1, (int)SZ_IN);
    prepack_kernel<<<(int)(SZ_HID / 256), 256, 0, stream>>>(tWhid,           p_tWh0,  HID,  4, (int)SZ_HID);
    prepack_kernel<<<(int)(SZ_HID / 256), 256, 0, stream>>>(tWhid + SZ_HID,  p_tWh1,  HID,  4, (int)SZ_HID);
    prepack_kernel<<<(int)(SZ_OUT / 256), 256, 0, stream>>>(tWout,           p_tWout, HID,  4, (int)SZ_OUT);

    affine_coupling_kernel<true><<<grid, 32, 0, stream>>>(
        z, p_sWin, sbin, p_sWh0, p_sWh1, sbhid, p_sWout, sbout,
        p_tWin, tbin, p_tWh0, p_tWh1, tbhid, p_tWout, tbout, out, logdet);
  } else {
    affine_coupling_kernel<false><<<grid, 32, 0, stream>>>(
        z, sWin, sbin, sWhid, sWhid + SZ_HID, sbhid, sWout, sbout,
        tWin, tbin, tWhid, tWhid + SZ_HID, tbhid, tWout, tbout, out, logdet);
  }
}
